// SDARAttention_22282290332156
// MI455X (gfx1250) — matrix operand order repaired and checked by reading, not compiled, not run on hardware
//
#include <hip/hip_runtime.h>

// ---------------- problem constants ----------------
#define B_      2
#define S_      2048
#define HID_    2048
#define HQ_     16
#define HKV_    4
#define D_      128
#define G_      4
#define WINDOW_ 1024
#define NQKV_   3072            // (HQ + 2*HKV) * D
#define MROWS_  4096            // B * S
#define SCALE_  0.08838834764831845f   // 1/sqrt(128)

typedef __attribute__((ext_vector_type(16))) __bf16 v16bf;
typedef __attribute__((ext_vector_type(8)))  float  v8f;
typedef __attribute__((ext_vector_type(4)))  unsigned v4u;
typedef __attribute__((ext_vector_type(8)))  unsigned v8u;
typedef unsigned short u16;

union FragB16 { v16bf v; uint4 q[2]; };
union FragF32 { v8f  v; float f[8]; };

__device__ __forceinline__ u16 f2bf(float x) {
  unsigned u = __float_as_uint(x);
  u += 0x7fffu + ((u >> 16) & 1u);      // round-to-nearest-even
  return (u16)(u >> 16);
}

// byte offset of a __shared__ object inside the workgroup's LDS allocation
__device__ __forceinline__ unsigned lds_addr_of(const void* p) {
  return (unsigned)(uintptr_t)(__attribute__((address_space(3))) const void*)p;
}

// ---------------- fp32 -> bf16 conversion (x4 vectorized, no transpose) ----------------
__global__ __launch_bounds__(256)
void cvt_f32_bf16(const float4* __restrict__ src, ushort4* __restrict__ dst, int n4) {
  int i = blockIdx.x * 256 + threadIdx.x;
  if (i < n4) {
    float4 f = src[i];
    ushort4 o;
    o.x = f2bf(f.x); o.y = f2bf(f.y); o.z = f2bf(f.z); o.w = f2bf(f.w);
    dst[i] = o;
  }
}

// ---------------- fp32 [K][N] -> bf16 [N][K] transpose-convert (weights, done once) ----------------
__global__ __launch_bounds__(256)
void cvt_transpose_bf16(const float* __restrict__ src, u16* __restrict__ dst,
                        int K, int N) {
  __shared__ u16 tile[32][33];          // +1 pad: conflict-free transpose
  const int k0 = blockIdx.y * 32;
  const int n0 = blockIdx.x * 32;
  const int tx = threadIdx.x & 31;
  const int ty = threadIdx.x >> 5;      // 8 rows per pass
#pragma unroll
  for (int r = ty; r < 32; r += 8)
    tile[r][tx] = f2bf(src[(size_t)(k0 + r) * N + n0 + tx]);
  __syncthreads();
#pragma unroll
  for (int r = ty; r < 32; r += 8)
    dst[(size_t)(n0 + r) * K + k0 + tx] = tile[tx][r];
}

// ---------------- bf16 WMMA GEMM: C(f32, MxN) = A(bf16, MxK) * Bt(bf16, NxK) ----------------
// block tile 128x128, BK=32, 256 threads (8 waves), wave tile 32x64 (2x4 wmma).
// Double-buffered: A tile staged by the Tensor Data Mover, B tile by async
// global->LDS b128 loads (B is pre-transposed, so it lands fragment-ready).
__global__ __launch_bounds__(256)
void wmma_gemm_bf16(const u16* __restrict__ A, const u16* __restrict__ Bt,
                    float* __restrict__ C, int M, int N, int K) {
  __shared__ u16 As[2][128][32];   // [m][k], TDM destination (8KB per buffer)
  __shared__ u16 Bs[2][128][32];   // [n][k], async-load destination
  const int tid  = threadIdx.x;
  const int lane = tid & 31;
  const int wid  = tid >> 5;
  const int bm   = blockIdx.y * 128;
  const int bn   = blockIdx.x * 128;
  const int wm   = (wid >> 1) * 32;
  const int wn   = (wid & 1) * 64;
  const int lm   = lane & 15;
  const int kh   = lane >> 4;

  v8f acc[2][4] = {};

  const int s_row  = tid >> 1;          // 0..127 : B row staged by this thread
  const int s_half = (tid & 1) << 4;    // 0 / 16 : element offset within row
  const unsigned as_lds = lds_addr_of(&As[0][0][0]);
  const unsigned bs_lds = lds_addr_of(&Bs[0][s_row][s_half]);

  auto stage = [&](int buf, int k0) {
    // A tile (128x32) : one TDM descriptor from wave 0
    if (wid == 0) {
      uint64_t ga = (uint64_t)(uintptr_t)(A + (size_t)bm * K + k0);
      v4u g0; v8u g1;
      g0[0] = 1u;                                    // count=1, user descriptor
      g0[1] = as_lds + (unsigned)buf * 8192u;        // lds_addr (bytes)
      g0[2] = (unsigned)ga;                          // global_addr[31:0]
      g0[3] = ((unsigned)(ga >> 32) & 0x01ffffffu)   // global_addr[56:32]
              | (2u << 30);                          // type = 2 ("image")
      g1[0] = 1u << 16;                              // data_size = 1 -> 2 bytes
      g1[1] = 32u << 16;                             // tensor_dim0 = 32
      g1[2] = 128u << 16;                            // tensor_dim1 = 128
      g1[3] = 32u << 16;                             // tile_dim0 = 32
      g1[4] = 128u;                                  // tile_dim1 = 128
      g1[5] = (unsigned)K;                           // tensor_dim0_stride (elems)
      g1[6] = 0u;
      g1[7] = 0u;
      asm volatile("tensor_load_to_lds %0, %1" :: "s"(g0), "s"(g1) : "memory");
    }
    // B tile (128 rows x 32 k, fragment-ready): 2 x 16B async per thread
    uint64_t gb = (uint64_t)(uintptr_t)(Bt + (size_t)(bn + s_row) * K + k0 + s_half);
    unsigned  ld = bs_lds + (unsigned)buf * 8192u;
    asm volatile("global_load_async_to_lds_b128 %0, %1, off"
                 :: "v"(ld), "v"(gb) : "memory");
    asm volatile("global_load_async_to_lds_b128 %0, %1, off"
                 :: "v"(ld + 16u), "v"(gb + 16u) : "memory");
  };

  stage(0, 0);
  for (int k0 = 0; k0 < K; k0 += 32) {
    const int buf = (k0 >> 5) & 1;
    asm volatile("s_wait_asynccnt 0x0" ::: "memory");
    if (wid == 0) __builtin_amdgcn_s_wait_tensorcnt(0);
    __syncthreads();
    if (k0 + 32 < K) stage(buf ^ 1, k0 + 32);
    if (k0 + 64 < K)   // keep the tile after next warm in GL2
      __builtin_prefetch(Bt + (size_t)(bn + s_row) * K + k0 + 64 + s_half, 0, 1);

    FragB16 af[2], bfb[4];
#pragma unroll
    for (int mi = 0; mi < 2; ++mi) {
      const u16* p = &As[buf][wm + mi * 16 + lm][kh * 8];
      af[mi].q[0] = *(const uint4*)p;
      af[mi].q[1] = *(const uint4*)(p + 16);
    }
#pragma unroll
    for (int ni = 0; ni < 4; ++ni) {
      const u16* p = &Bs[buf][wn + ni * 16 + lm][kh * 8];
      bfb[ni].q[0] = *(const uint4*)p;
      bfb[ni].q[1] = *(const uint4*)(p + 16);
    }
#pragma unroll
    for (int mi = 0; mi < 2; ++mi)
#pragma unroll
      for (int ni = 0; ni < 4; ++ni)
        acc[mi][ni] = __builtin_amdgcn_wmma_f32_16x16x32_bf16(
            false, af[mi].v, false, bfb[ni].v, (short)0, acc[mi][ni], false, false);
  }

#pragma unroll
  for (int mi = 0; mi < 2; ++mi)
#pragma unroll
    for (int ni = 0; ni < 4; ++ni) {
      FragF32 r; r.v = acc[mi][ni];
#pragma unroll
      for (int rr = 0; rr < 8; ++rr) {
        int m = bm + wm + mi * 16 + rr + 8 * kh;
        int n = bn + wn + ni * 16 + lm;
        C[(size_t)m * N + n] = r.f[rr];
      }
    }
}

// ---------------- RMSNorm + RoPE epilogue, scatter to Q/K/Vt (bf16) ----------------
__global__ __launch_bounds__(128)
void norm_rope_scatter(const float* __restrict__ qkv,
                       const float* __restrict__ cosb, const float* __restrict__ sinb,
                       const float* __restrict__ qw,   const float* __restrict__ kw,
                       u16* __restrict__ Qb, u16* __restrict__ Kb, u16* __restrict__ Vt) {
  const int bs   = blockIdx.x;
  const int b    = bs / S_;
  const int s    = bs % S_;
  const int tid  = threadIdx.x;
  const int lane = tid & 31;
  const int w    = tid >> 5;
  const float* row = qkv + (size_t)bs * NQKV_;

  for (int h = w; h < HQ_; h += 4) {
    const float* x = row + h * D_;
    float ss = 0.f;
#pragma unroll
    for (int t = 0; t < 4; ++t) { float v = x[lane + t * 32]; ss += v * v; }
#pragma unroll
    for (int m = 16; m; m >>= 1) ss += __shfl_xor(ss, m, 32);
    float rstd = rsqrtf(ss * (1.f / 128.f) + 1e-6f);
    u16* dst = Qb + ((size_t)(b * HQ_ + h) * S_ + s) * D_;
#pragma unroll
    for (int t = 0; t < 4; ++t) {
      int d = lane + t * 32;
      float xn = x[d] * rstd * qw[d];
      int   dp = (d < 64) ? d + 64 : d - 64;
      float pn = x[dp] * rstd * qw[dp];
      float rh = (d < 64) ? -pn : pn;
      dst[d] = f2bf(xn * cosb[s * D_ + d] + rh * sinb[s * D_ + d]);
    }
  }
  {
    const int h = w;
    const float* x = row + HQ_ * D_ + h * D_;
    float ss = 0.f;
#pragma unroll
    for (int t = 0; t < 4; ++t) { float v = x[lane + t * 32]; ss += v * v; }
#pragma unroll
    for (int m = 16; m; m >>= 1) ss += __shfl_xor(ss, m, 32);
    float rstd = rsqrtf(ss * (1.f / 128.f) + 1e-6f);
    u16* dst = Kb + ((size_t)(b * HKV_ + h) * S_ + s) * D_;
#pragma unroll
    for (int t = 0; t < 4; ++t) {
      int d = lane + t * 32;
      float xn = x[d] * rstd * kw[d];
      int   dp = (d < 64) ? d + 64 : d - 64;
      float pn = x[dp] * rstd * kw[dp];
      float rh = (d < 64) ? -pn : pn;
      dst[d] = f2bf(xn * cosb[s * D_ + d] + rh * sinb[s * D_ + d]);
    }
  }
  for (int idx = tid; idx < HKV_ * D_; idx += 128) {
    int h = idx >> 7, d = idx & 127;
    float v = row[(HQ_ + HKV_) * D_ + h * D_ + d];
    Vt[((size_t)(b * HKV_ + h) * D_ + d) * S_ + s] = f2bf(v);
  }
}

// ---------------- sliding-window flash attention (WMMA) ----------------
__global__ __launch_bounds__(128)
void attn_swa(const u16* __restrict__ Qb, const u16* __restrict__ Kb,
              const u16* __restrict__ Vt, u16* __restrict__ Oa) {
  __shared__ u16 Pt[4][16][32];
  const int tid  = threadIdx.x;
  const int lane = tid & 31;
  const int wid  = tid >> 5;
  const int gid  = blockIdx.x * 4 + wid;
  const int qb   = gid & 127;
  const int h    = (gid >> 7) & 15;
  const int b    = gid >> 11;
  const int kv   = h >> 2;
  const int q0   = qb << 4;
  const int lm   = lane & 15;
  const int kh   = lane >> 4;

  const u16* Qp = Qb + (size_t)(b * HQ_  + h)  * S_ * D_;
  const u16* Kp = Kb + (size_t)(b * HKV_ + kv) * S_ * D_;
  const u16* Vp = Vt + (size_t)(b * HKV_ + kv) * D_ * S_;

  FragB16 qf[4];
#pragma unroll
  for (int kk = 0; kk < 4; ++kk) {
    const u16* p = Qp + (size_t)(q0 + lm) * D_ + kk * 32 + kh * 8;
    qf[kk].q[0] = *(const uint4*)p;
    qf[kk].q[1] = *(const uint4*)(p + 16);
  }

  float mrow[8], lrow[8];
#pragma unroll
  for (int r = 0; r < 8; ++r) { mrow[r] = -1e30f; lrow[r] = 0.f; }
  v8f o[8] = {};

  int j_lo = q0 + 15 - (WINDOW_ - 1);
  if (j_lo < 0) j_lo = 0;
  j_lo &= ~31;

  for (int jb = j_lo; jb <= q0 + 15; jb += 32) {
    v8f s0 = {}, s1 = {};
#pragma unroll
    for (int kk = 0; kk < 4; ++kk) {
      FragB16 kf;
      const u16* p = Kp + (size_t)(jb + lm) * D_ + kk * 32 + kh * 8;
      kf.q[0] = *(const uint4*)p;
      kf.q[1] = *(const uint4*)(p + 16);
      s0 = __builtin_amdgcn_wmma_f32_16x16x32_bf16(false, qf[kk].v, false, kf.v,
                                                   (short)0, s0, false, false);
      const u16* p2 = Kp + (size_t)(jb + 16 + lm) * D_ + kk * 32 + kh * 8;
      kf.q[0] = *(const uint4*)p2;
      kf.q[1] = *(const uint4*)(p2 + 16);
      s1 = __builtin_amdgcn_wmma_f32_16x16x32_bf16(false, qf[kk].v, false, kf.v,
                                                   (short)0, s1, false, false);
    }
    FragF32 sf0, sf1; sf0.v = s0; sf1.v = s1;
    float p0[8], p1[8];
#pragma unroll
    for (int r = 0; r < 8; ++r) {
      const int mq = q0 + r + 8 * kh;
      const int j0 = jb + lm, j1 = jb + 16 + lm;
      float v0 = sf0.f[r] * SCALE_;
      float v1 = sf1.f[r] * SCALE_;
      if (!(j0 <= mq && mq - j0 < WINDOW_)) v0 = -1e30f;
      if (!(j1 <= mq && mq - j1 < WINDOW_)) v1 = -1e30f;
      float rmax = fmaxf(v0, v1);
#pragma unroll
      for (int m2 = 1; m2 < 16; m2 <<= 1) rmax = fmaxf(rmax, __shfl_xor(rmax, m2, 32));
      float mnew  = fmaxf(mrow[r], rmax);
      float alpha = __expf(mrow[r] - mnew);
      float e0 = __expf(v0 - mnew);
      float e1 = __expf(v1 - mnew);
      float rs = e0 + e1;
#pragma unroll
      for (int m2 = 1; m2 < 16; m2 <<= 1) rs += __shfl_xor(rs, m2, 32);
      lrow[r] = lrow[r] * alpha + rs;
      mrow[r] = mnew;
      p0[r] = e0; p1[r] = e1;
#pragma unroll
      for (int dt = 0; dt < 8; ++dt) o[dt][r] *= alpha;
    }
#pragma unroll
    for (int r = 0; r < 8; ++r) {
      const int m = r + 8 * kh;
      Pt[wid][m][lm]      = f2bf(p0[r]);
      Pt[wid][m][16 + lm] = f2bf(p1[r]);
    }
    FragB16 pf;
    {
      const u16* p = &Pt[wid][lm][kh * 8];
      pf.q[0] = *(const uint4*)p;
      pf.q[1] = *(const uint4*)(p + 16);
    }
#pragma unroll
    for (int dt = 0; dt < 8; ++dt) {
      FragB16 vf;
      const u16* p = Vp + (size_t)(dt * 16 + lm) * S_ + jb + kh * 8;
      vf.q[0] = *(const uint4*)p;
      vf.q[1] = *(const uint4*)(p + 16);
      o[dt] = __builtin_amdgcn_wmma_f32_16x16x32_bf16(false, pf.v, false, vf.v,
                                                      (short)0, o[dt], false, false);
    }
  }

#pragma unroll
  for (int dt = 0; dt < 8; ++dt) {
    FragF32 of; of.v = o[dt];
#pragma unroll
    for (int r = 0; r < 8; ++r) {
      const int m = r + 8 * kh;
      float val = of.f[r] / lrow[r];
      Oa[((size_t)b * S_ + q0 + m) * (HQ_ * D_) + h * D_ + dt * 16 + lm] = f2bf(val);
    }
  }
}

// ---------------- host: launch pipeline ----------------
extern "C" void kernel_launch(void* const* d_in, const int* in_sizes, int n_in,
                              void* d_out, int out_size, void* d_ws, size_t ws_size,
                              hipStream_t stream) {
  (void)in_sizes; (void)n_in; (void)out_size; (void)ws_size;
  const float* hs   = (const float*)d_in[0];
  const float* cosb = (const float*)d_in[1];
  const float* sinb = (const float*)d_in[2];
  const float* wqkv = (const float*)d_in[3];
  const float* wo   = (const float*)d_in[4];
  const float* qw   = (const float*)d_in[5];
  const float* kw   = (const float*)d_in[6];
  float* out = (float*)d_out;

  char* ws = (char*)d_ws;
  size_t off = 0;
  auto carve = [&](size_t bytes) -> void* {
    void* p = ws + off;
    off = (off + bytes + 255) & ~(size_t)255;
    return p;
  };
  u16*   hs_bf    = (u16*)  carve((size_t)MROWS_ * HID_  * 2);
  u16*   wqkv_t   = (u16*)  carve((size_t)HID_   * NQKV_ * 2);  // [N][K]
  u16*   wo_t     = (u16*)  carve((size_t)HID_   * HID_  * 2);  // [N][K]
  float* qkv_f    = (float*)carve((size_t)MROWS_ * NQKV_ * 4);
  u16*   Qbf      = (u16*)  carve((size_t)B_ * HQ_  * S_ * D_ * 2);
  u16*   Kbf      = (u16*)  carve((size_t)B_ * HKV_ * S_ * D_ * 2);
  u16*   Vtbf     = (u16*)  carve((size_t)B_ * HKV_ * D_ * S_ * 2);
  u16*   attn_bf  = (u16*)  carve((size_t)MROWS_ * HID_ * 2);

  // activations: straight convert; weights: transpose-convert (once)
  {
    int n4 = (MROWS_ * HID_) / 4;
    cvt_f32_bf16<<<(n4 + 255) / 256, 256, 0, stream>>>(
        (const float4*)hs, (ushort4*)hs_bf, n4);
  }
  cvt_transpose_bf16<<<dim3(NQKV_ / 32, HID_ / 32), 256, 0, stream>>>(
      wqkv, wqkv_t, HID_, NQKV_);
  cvt_transpose_bf16<<<dim3(HID_ / 32, HID_ / 32), 256, 0, stream>>>(
      wo, wo_t, HID_, HID_);

  // QKV projection: (4096 x 2048) * (2048 x 3072)
  wmma_gemm_bf16<<<dim3(NQKV_ / 128, MROWS_ / 128), 256, 0, stream>>>(
      hs_bf, wqkv_t, qkv_f, MROWS_, NQKV_, HID_);
  // RMSNorm + RoPE + scatter
  norm_rope_scatter<<<B_ * S_, 128, 0, stream>>>(
      qkv_f, cosb, sinb, qw, kw, Qbf, Kbf, Vtbf);
  // sliding-window attention
  attn_swa<<<(B_ * HQ_ * (S_ / 16)) / 4, 128, 0, stream>>>(Qbf, Kbf, Vtbf, attn_bf);
  // output projection: (4096 x 2048) * (2048 x 2048)
  wmma_gemm_bf16<<<dim3(HID_ / 128, MROWS_ / 128), 256, 0, stream>>>(
      attn_bf, wo_t, out, MROWS_, HID_, HID_);
}
